// NodeLevelAttention_73057393705159
// MI455X (gfx1250) — matrix operand order repaired and checked by reading, not compiled, not run on hardware
//
#include <hip/hip_runtime.h>
#include <math.h>

#define IN_DIM 128
#define NHEAD 4
#define CDIM 64
#define HC 256          // NHEAD * CDIM
#define NEG_SLOPE 0.2f

typedef __attribute__((ext_vector_type(16))) _Float16 v16h;
typedef __attribute__((ext_vector_type(8)))  float    v8f;

// ---------------------------------------------------------------------------
// K0: initialize workspace (emax = -inf, denom = 0, accum = 0)
// ---------------------------------------------------------------------------
__global__ void gat_init(float* __restrict__ emax, float* __restrict__ denom,
                         float* __restrict__ accum, int N) {
    int idx = blockIdx.x * blockDim.x + threadIdx.x;
    int nh = N * NHEAD;
    if (idx < nh) {
        emax[idx]  = __int_as_float(0xFF800000); // -inf (works with bit-trick atomics)
        denom[idx] = 0.0f;
    }
    if (idx < N * HC) accum[idx] = 0.0f;
}

// ---------------------------------------------------------------------------
// K0b: pack W (f32 row-major [K, HC]) into f16 B-fragment layout:
//   Wp[(((ntile*4 + chunk)*32 + lane)*16 + i] = W[(chunk*32 + (i>>3)*16 + (lane>>4)*8 + (i&7))*HC
//                                                 + ntile*16 + (lane&15)]
// 64 KB total -> L0/L2 resident; GEMM B load becomes one 32B vector load.
// ---------------------------------------------------------------------------
__global__ void gat_pack_w(const float* __restrict__ W, _Float16* __restrict__ Wp) {
    int idx = blockIdx.x * blockDim.x + threadIdx.x;
    if (idx >= IN_DIM * HC) return;
    int i     = idx & 15;
    int lane  = (idx >> 4) & 31;
    int chunk = (idx >> 9) & 3;
    int ntile = idx >> 11;
    int K   = chunk * 32 + (i >> 3) * 16 + (lane >> 4) * 8 + (i & 7);
    int col = ntile * 16 + (lane & 15);
    Wp[idx] = (_Float16)W[(size_t)K * HC + col];
}

// ---------------------------------------------------------------------------
// K1: h = x @ W via v_wmma_f32_16x16x32_f16.
// One wave computes a 16x64 output strip (4 N-tiles) sharing one A fragment.
// ---------------------------------------------------------------------------
__global__ void gat_gemm_wmma(const float* __restrict__ x,
                              const _Float16* __restrict__ Wp,
                              float* __restrict__ h, int N) {
    int wave   = (blockIdx.x * blockDim.x + threadIdx.x) >> 5;
    int lane   = threadIdx.x & 31;
    int mtiles = (N + 15) >> 4;
    int ngrp   = wave & 3;           // 4 groups of 4 col-tiles cover HC=256
    int mtile  = wave >> 2;
    if (mtile >= mtiles) return;     // wave-uniform branch, EXEC stays all-1

    int lhalf = lane >> 4;           // 0 or 1
    int lmod  = lane & 15;
    int row   = (mtile << 4) + lmod;
    if (row >= N) row = N - 1;       // clamp (keeps all lanes active for WMMA)

    v8f c0 = {}, c1 = {}, c2 = {}, c3 = {};
    #pragma unroll
    for (int k0 = 0; k0 < IN_DIM; k0 += 32) {
        // A fragment: 16-bit A 16x32 layout; pairs v=0..3 -> K in [k0,k0+16),
        // v=4..7 -> [k0+16,k0+32), lane half adds +8.
        v16h a;
        const float* xr = x + (size_t)row * IN_DIM + k0 + lhalf * 8;
        #pragma unroll
        for (int v = 0; v < 4; ++v) {
            float2 f = *(const float2*)(xr + 2 * v);
            a[2 * v]         = (_Float16)f.x;
            a[2 * v + 1]     = (_Float16)f.y;
            float2 g = *(const float2*)(xr + 16 + 2 * v);
            a[8 + 2 * v]     = (_Float16)g.x;
            a[8 + 2 * v + 1] = (_Float16)g.y;
        }
        int chunk = k0 >> 5;
        #pragma unroll
        for (int t = 0; t < 4; ++t) {
            int ntile = ngrp * 4 + t;
            const v16h* bp = (const v16h*)(Wp + (size_t)(((ntile * 4 + chunk) * 32
                                                          + lane) << 4));
            v16h b = *bp;
            v8f* cp = (t == 0) ? &c0 : (t == 1) ? &c1 : (t == 2) ? &c2 : &c3;
            *cp = __builtin_amdgcn_wmma_f32_16x16x32_f16(
                    false, a, false, b, (short)0, *cp, false, false);
        }
    }
    // C/D: VGPR r holds rows r (lanes 0-15) and r+8 (lanes 16-31)
    int rbase = (mtile << 4) + (lhalf << 3);
    #pragma unroll
    for (int t = 0; t < 4; ++t) {
        int col = (ngrp * 4 + t) * 16 + lmod;
        const v8f& c = (t == 0) ? c0 : (t == 1) ? c1 : (t == 2) ? c2 : c3;
        #pragma unroll
        for (int r = 0; r < 8; ++r) {
            int orow = rbase + r;
            if (orow < N) h[(size_t)orow * HC + col] = c[r];
        }
    }
}

// ---------------------------------------------------------------------------
// K2: per-node attention terms  a_src[n,h] = <h[n,h,:], att_src[h,:]>, same dst
// ---------------------------------------------------------------------------
__global__ void gat_att(const float* __restrict__ h,
                        const float* __restrict__ att_src,
                        const float* __restrict__ att_dst,
                        float* __restrict__ a_src, float* __restrict__ a_dst, int N) {
    int idx = blockIdx.x * blockDim.x + threadIdx.x;
    if (idx >= N * NHEAD) return;
    int n = idx >> 2, hh = idx & 3;
    const float4* hp = (const float4*)(h + (size_t)n * HC + hh * CDIM);
    const float4* as = (const float4*)(att_src + hh * CDIM);
    const float4* ad = (const float4*)(att_dst + hh * CDIM);
    float s = 0.f, d = 0.f;
    #pragma unroll
    for (int q = 0; q < CDIM / 4; ++q) {
        float4 v = hp[q], a = as[q], b = ad[q];
        s += v.x * a.x + v.y * a.y + v.z * a.z + v.w * a.w;
        d += v.x * b.x + v.y * b.y + v.z * b.z + v.w * b.w;
    }
    a_src[idx] = s;
    a_dst[idx] = d;
}

// float atomic-max via sign-split int/uint atomics (no CAS loop)
__device__ __forceinline__ void atomicMaxF32(float* addr, float val) {
    if (val >= 0.f) atomicMax((int*)addr, __float_as_int(val));
    else            atomicMin((unsigned int*)addr, __float_as_uint(val));
}

__device__ __forceinline__ void edge_ids(const int* __restrict__ ei, int idx, int E,
                                         int& s, int& d) {
    if (idx < E) { s = ei[idx]; d = ei[E + idx]; }
    else         { s = idx - E; d = idx - E; }     // self loop
}

// ---------------------------------------------------------------------------
// K3: segment max of leaky-relu logits
// ---------------------------------------------------------------------------
__global__ void gat_edge_max(const int* __restrict__ ei,
                             const float* __restrict__ a_src,
                             const float* __restrict__ a_dst,
                             float* __restrict__ emax, int E, int N) {
    int idx = blockIdx.x * blockDim.x + threadIdx.x;
    if (idx >= E + N) return;
    int s, d; edge_ids(ei, idx, E, s, d);
    float4 as = *(const float4*)(a_src + (size_t)s * 4);
    float4 ad = *(const float4*)(a_dst + (size_t)d * 4);
    float e[4] = {as.x + ad.x, as.y + ad.y, as.z + ad.z, as.w + ad.w};
    #pragma unroll
    for (int k = 0; k < 4; ++k) {
        float v = e[k] > 0.f ? e[k] : NEG_SLOPE * e[k];
        atomicMaxF32(&emax[(size_t)d * 4 + k], v);
    }
}

// ---------------------------------------------------------------------------
// K4: segment sum of exp(e - emax)
// ---------------------------------------------------------------------------
__global__ void gat_edge_sum(const int* __restrict__ ei,
                             const float* __restrict__ a_src,
                             const float* __restrict__ a_dst,
                             const float* __restrict__ emax,
                             float* __restrict__ denom, int E, int N) {
    int idx = blockIdx.x * blockDim.x + threadIdx.x;
    if (idx >= E + N) return;
    int s, d; edge_ids(ei, idx, E, s, d);
    float4 as = *(const float4*)(a_src + (size_t)s * 4);
    float4 ad = *(const float4*)(a_dst + (size_t)d * 4);
    float4 mx = *(const float4*)(emax + (size_t)d * 4);
    float e[4] = {as.x + ad.x, as.y + ad.y, as.z + ad.z, as.w + ad.w};
    float m[4] = {mx.x, mx.y, mx.z, mx.w};
    #pragma unroll
    for (int k = 0; k < 4; ++k) {
        float v = e[k] > 0.f ? e[k] : NEG_SLOPE * e[k];
        atomicAdd(&denom[(size_t)d * 4 + k], expf(v - m[k]));
    }
}

// ---------------------------------------------------------------------------
// K5: weighted scatter-aggregate, one wave32 per edge.
// Gather h[src] with 2 coalesced float4 loads per lane; 8 f32 atomics per lane.
// ---------------------------------------------------------------------------
__global__ void gat_edge_aggr(const int* __restrict__ ei,
                              const float* __restrict__ a_src,
                              const float* __restrict__ a_dst,
                              const float* __restrict__ emax,
                              const float* __restrict__ denom,
                              const float* __restrict__ h,
                              float* __restrict__ accum, int E, int N) {
    int wave = (blockIdx.x * blockDim.x + threadIdx.x) >> 5;
    int lane = threadIdx.x & 31;
    if (wave >= E + N) return;
    int s, d; edge_ids(ei, wave, E, s, d);
    float4 as = *(const float4*)(a_src + (size_t)s * 4);
    float4 ad = *(const float4*)(a_dst + (size_t)d * 4);
    float4 mx = *(const float4*)(emax + (size_t)d * 4);
    float4 dn = *(const float4*)(denom + (size_t)d * 4);
    float e[4] = {as.x + ad.x, as.y + ad.y, as.z + ad.z, as.w + ad.w};
    float m[4] = {mx.x, mx.y, mx.z, mx.w};
    float q[4] = {dn.x, dn.y, dn.z, dn.w};
    float alpha[4];
    #pragma unroll
    for (int k = 0; k < 4; ++k) {
        float v = e[k] > 0.f ? e[k] : NEG_SLOPE * e[k];
        alpha[k] = expf(v - m[k]) / q[k];
    }
    const float4* hs4 = (const float4*)(h + (size_t)s * HC);
    float* ap = accum + (size_t)d * HC;
    #pragma unroll
    for (int j = 0; j < 2; ++j) {
        int base = j * 128 + lane * 4;      // 4 consecutive floats, same head block
        float4 hv = hs4[j * 32 + lane];
        float al = alpha[base >> 6];
        atomicAdd(&ap[base + 0], al * hv.x);
        atomicAdd(&ap[base + 1], al * hv.y);
        atomicAdd(&ap[base + 2], al * hv.z);
        atomicAdd(&ap[base + 3], al * hv.w);
    }
}

// ---------------------------------------------------------------------------
// K6: mean over heads + bias + ELU
// ---------------------------------------------------------------------------
__global__ void gat_finalize(const float* __restrict__ accum,
                             const float* __restrict__ bias,
                             float* __restrict__ out, int N) {
    int idx = blockIdx.x * blockDim.x + threadIdx.x;
    if (idx >= N * CDIM) return;
    int n = idx >> 6, cidx = idx & 63;
    const float* ap = accum + (size_t)n * HC + cidx;
    float v = 0.25f * (ap[0] + ap[64] + ap[128] + ap[192]) + bias[cidx];
    out[idx] = v > 0.f ? v : expf(v) - 1.f;
}

// ---------------------------------------------------------------------------
extern "C" void kernel_launch(void* const* d_in, const int* in_sizes, int n_in,
                              void* d_out, int out_size, void* d_ws, size_t ws_size,
                              hipStream_t stream) {
    const float* x       = (const float*)d_in[0];
    const int*   ei      = (const int*)d_in[1];
    const float* W       = (const float*)d_in[2];
    const float* att_src = (const float*)d_in[3];
    const float* att_dst = (const float*)d_in[4];
    const float* bias    = (const float*)d_in[5];
    float*       out     = (float*)d_out;

    const int N = in_sizes[0] / IN_DIM;
    const int E = in_sizes[1] / 2;

    float* ws     = (float*)d_ws;
    float* h      = ws;                              // N*HC
    float* a_src  = h     + (size_t)N * HC;          // N*NHEAD
    float* a_dst  = a_src + (size_t)N * NHEAD;       // N*NHEAD
    float* emax   = a_dst + (size_t)N * NHEAD;       // N*NHEAD
    float* denom  = emax  + (size_t)N * NHEAD;       // N*NHEAD
    float* accum  = denom + (size_t)N * NHEAD;       // N*HC
    // packed f16 W after accum, aligned up to 32 bytes
    uintptr_t wp_addr = (uintptr_t)(accum + (size_t)N * HC);
    wp_addr = (wp_addr + 31u) & ~(uintptr_t)31u;
    _Float16* Wp = (_Float16*)wp_addr;               // IN_DIM*HC halves (64 KB)

    const int TPB = 256;

    // K0: init
    gat_init<<<(N * HC + TPB - 1) / TPB, TPB, 0, stream>>>(emax, denom, accum, N);
    // K0b: pack W into f16 B-fragment layout
    gat_pack_w<<<(IN_DIM * HC + TPB - 1) / TPB, TPB, 0, stream>>>(W, Wp);

    // K1: WMMA GEMM (one wave per 16x64 strip; 8 waves per block)
    {
        int mtiles = (N + 15) / 16;
        int waves  = mtiles * 4;
        gat_gemm_wmma<<<(waves + 7) / 8, TPB, 0, stream>>>(x, Wp, h, N);
    }

    // K2: attention terms
    gat_att<<<(N * NHEAD + TPB - 1) / TPB, TPB, 0, stream>>>(h, att_src, att_dst,
                                                             a_src, a_dst, N);

    const int ETOT = E + N;
    // K3: segment max
    gat_edge_max<<<(ETOT + TPB - 1) / TPB, TPB, 0, stream>>>(ei, a_src, a_dst, emax, E, N);
    // K4: segment sum
    gat_edge_sum<<<(ETOT + TPB - 1) / TPB, TPB, 0, stream>>>(ei, a_src, a_dst, emax,
                                                             denom, E, N);
    // K5: aggregate (wave per edge)
    gat_edge_aggr<<<(ETOT + 7) / 8, TPB, 0, stream>>>(ei, a_src, a_dst, emax, denom,
                                                      h, accum, E, N);
    // K6: finalize
    gat_finalize<<<(N * CDIM + TPB - 1) / TPB, TPB, 0, stream>>>(accum, bias, out, N);
}
